// MultiHeadAttention_12884901888214
// MI455X (gfx1250) — compile-verified, matrix operand order checked by reading, not run on hardware
//
#include <hip/hip_runtime.h>
#include <hip/hip_bf16.h>

typedef __bf16 bf16;
typedef __attribute__((ext_vector_type(16))) __bf16 v16bf;
typedef __attribute__((ext_vector_type(8)))  float  v8f;

#define D_MODEL 1024
#define N_HEADS 16
#define DK      64
#define BATCH   2
#define SEQ     2048
#define M_TOK   (BATCH * SEQ)   // 4096

// ---- CDNA5 async copy global -> LDS (ASYNCcnt-tracked) ---------------------
__device__ __forceinline__ void async_b128(void* lds_ptr, const void* gptr) {
    // VDST = LDS byte address (low 32 bits of the generic shared pointer),
    // VADDR = 64-bit global address, no SADDR.
    unsigned l = (unsigned)(unsigned long long)lds_ptr;
    asm volatile("global_load_async_to_lds_b128 %0, %1, off"
                 :: "v"(l), "v"(gptr) : "memory");
}
__device__ __forceinline__ void wait_async_0() {
    asm volatile("s_wait_asynccnt 0x0" ::: "memory");
}
template <int N>
__device__ __forceinline__ void wait_async_le() {
    asm volatile("s_wait_asynccnt %0" :: "i"(N) : "memory");
}

// ---------------------------------------------------------------------------
// Dense projection GEMM: C[M=4096,N=1024] = A[M,1024] * W[N=1024,K=1024]^T + b
//   A_BF16 : A operand is bf16 workspace (async-to-LDS) vs fp32 (convert path)
//   MODE 0 : bf16 out, [B,H,S,dk]   (q, k)
//   MODE 1 : bf16 out, [B,H,dk,S]   (v transposed for the PV GEMM)
//   MODE 2 : fp32 out + bias, row-major [M,N] into d_out (final projection)
// Tile 128x128x32, double-buffered LDS, 8 waves as 4(M)x2(N), 2x4 frags/wave.
// ---------------------------------------------------------------------------
template <int A_BF16, int MODE>
__global__ __launch_bounds__(256)
void proj_gemm_kernel(const void* __restrict__ Aptr, const float* __restrict__ W,
                      const float* __restrict__ bias, void* __restrict__ Optr)
{
    constexpr int BM = 128, BN = 128, BK = 32, LDK = BK + 16;  // 96B row pitch
    __shared__ bf16 As[2][BM][LDK];
    __shared__ bf16 Bs[2][BN][LDK];

    const int m0 = blockIdx.y * BM;
    const int n0 = blockIdx.x * BN;
    const int t    = threadIdx.x;
    const int lane = t & 31;
    const int wave = t >> 5;
    const int waveM = wave >> 1;          // 0..3
    const int waveN = wave & 1;           // 0..1
    const int lr = lane & 15;
    const int lk = (lane >> 4) << 4;

    v8f acc[2][4] = {};

    const int arow  = t >> 1;             // 0..127
    const int akseg = (t & 1) << 4;       // 0 / 16

    auto stageA = [&](int k0, int buf) {
        if constexpr (A_BF16) {
            const bf16* A = (const bf16*)Aptr;
            const bf16* g = A + (size_t)(m0 + arow) * 1024 + k0 + akseg;
            async_b128(&As[buf][arow][akseg],     g);
            async_b128(&As[buf][arow][akseg + 8], g + 8);
        } else {
            const float* A = (const float*)Aptr;
            const float4* src = (const float4*)(A + (size_t)(m0 + arow) * 1024 + k0 + akseg);
#pragma unroll
            for (int q = 0; q < 4; ++q) {
                float4 x = src[q];
                As[buf][arow][akseg + q * 4 + 0] = (bf16)x.x;
                As[buf][arow][akseg + q * 4 + 1] = (bf16)x.y;
                As[buf][arow][akseg + q * 4 + 2] = (bf16)x.z;
                As[buf][arow][akseg + q * 4 + 3] = (bf16)x.w;
            }
        }
    };
    auto stageB = [&](int k0, int buf) {   // weights are always fp32 -> convert
        const float4* src = (const float4*)(W + (size_t)(n0 + arow) * 1024 + k0 + akseg);
#pragma unroll
        for (int q = 0; q < 4; ++q) {
            float4 x = src[q];
            Bs[buf][arow][akseg + q * 4 + 0] = (bf16)x.x;
            Bs[buf][arow][akseg + q * 4 + 1] = (bf16)x.y;
            Bs[buf][arow][akseg + q * 4 + 2] = (bf16)x.z;
            Bs[buf][arow][akseg + q * 4 + 3] = (bf16)x.w;
        }
    };

    stageA(0, 0);
    stageB(0, 0);

    constexpr int NIT = 1024 / BK;        // 32
    for (int it = 0; it < NIT; ++it) {
        const int cur = it & 1;
        const bool hasNext = (it + 1) < NIT;
        if (hasNext) {
            stageA((it + 1) * BK, cur ^ 1);
            stageB((it + 1) * BK, cur ^ 1);
            __builtin_prefetch(W + (size_t)(n0 + arow) * 1024 + (it + 1) * BK + akseg, 0, 1);
        }
        if constexpr (A_BF16) {
            if (hasNext) wait_async_le<2>(); else wait_async_0();
        }
        __syncthreads();

        v16bf a[2], b[4];
#pragma unroll
        for (int f = 0; f < 2; ++f)
            a[f] = *(const v16bf*)&As[cur][waveM * 32 + f * 16 + lr][lk];
#pragma unroll
        for (int g = 0; g < 4; ++g)
            b[g] = *(const v16bf*)&Bs[cur][waveN * 64 + g * 16 + lr][lk];
#pragma unroll
        for (int f = 0; f < 2; ++f)
#pragma unroll
            for (int g = 0; g < 4; ++g)
                acc[f][g] = __builtin_amdgcn_wmma_f32_16x16x32_bf16(
                    false, a[f], false, b[g], (short)0, acc[f][g], false, false);
        __syncthreads();
    }

    const int nl = lane & 15;
    const int mh = lane >> 4;
#pragma unroll
    for (int f = 0; f < 2; ++f) {
#pragma unroll
        for (int g = 0; g < 4; ++g) {
            const int n = n0 + waveN * 64 + g * 16 + nl;
            const float bv = bias[n];
#pragma unroll
            for (int r = 0; r < 8; ++r) {
                const int m = m0 + waveM * 32 + f * 16 + r + 8 * mh;
                const float val = acc[f][g][r] + bv;
                if constexpr (MODE == 2) {
                    ((float*)Optr)[(size_t)m * 1024 + n] = val;
                } else {
                    const int b_ = m >> 11, s = m & 2047;   // batch, seq
                    const int h  = n >> 6,  d = n & 63;     // head, dk
                    size_t idx;
                    if constexpr (MODE == 0)
                        idx = (((size_t)b_ * N_HEADS + h) * SEQ + s) * DK + d;
                    else
                        idx = (((size_t)b_ * N_HEADS + h) * DK + d) * SEQ + s;
                    ((bf16*)Optr)[idx] = (bf16)val;
                }
            }
        }
    }
}

// ---------------------------------------------------------------------------
// scores[b,h] = (q @ k^T)/sqrt(dk) : M=N=2048, K=64 per (b,h); fp32 to d_out.
// Both operands bf16 -> fully async-to-LDS, double-buffered (2 K-steps).
// ---------------------------------------------------------------------------
__global__ __launch_bounds__(256)
void score_gemm_kernel(const bf16* __restrict__ qbf, const bf16* __restrict__ kbf,
                       float* __restrict__ attn)
{
    constexpr int BM = 128, BN = 128, BK = 32, LDK = BK + 16;
    __shared__ bf16 As[2][BM][LDK];
    __shared__ bf16 Bs[2][BN][LDK];

    const int bh = blockIdx.z;                       // 0..31
    const bf16* A  = qbf + (size_t)bh * SEQ * DK;
    const bf16* Bm = kbf + (size_t)bh * SEQ * DK;
    float* out = attn + (size_t)bh * SEQ * SEQ;

    const int m0 = blockIdx.y * BM;
    const int n0 = blockIdx.x * BN;
    const int t = threadIdx.x, lane = t & 31, wave = t >> 5;
    const int waveM = wave >> 1, waveN = wave & 1;
    const int lr = lane & 15, lk = (lane >> 4) << 4;

    v8f acc[2][4] = {};
    const int arow = t >> 1, akseg = (t & 1) << 4;

    auto stage = [&](int k0, int buf) {
        const bf16* ga = A  + (size_t)(m0 + arow) * DK + k0 + akseg;
        const bf16* gb = Bm + (size_t)(n0 + arow) * DK + k0 + akseg;
        async_b128(&As[buf][arow][akseg],     ga);
        async_b128(&As[buf][arow][akseg + 8], ga + 8);
        async_b128(&Bs[buf][arow][akseg],     gb);
        async_b128(&Bs[buf][arow][akseg + 8], gb + 8);
    };

    stage(0, 0);
    constexpr int NIT = DK / BK;                     // 2
#pragma unroll
    for (int it = 0; it < NIT; ++it) {
        const int cur = it & 1;
        const bool hasNext = (it + 1) < NIT;
        if (hasNext) stage((it + 1) * BK, cur ^ 1);
        if (hasNext) wait_async_le<4>(); else wait_async_0();
        __syncthreads();

        v16bf a[2], b[4];
#pragma unroll
        for (int f = 0; f < 2; ++f)
            a[f] = *(const v16bf*)&As[cur][waveM * 32 + f * 16 + lr][lk];
#pragma unroll
        for (int g = 0; g < 4; ++g)
            b[g] = *(const v16bf*)&Bs[cur][waveN * 64 + g * 16 + lr][lk];
#pragma unroll
        for (int f = 0; f < 2; ++f)
#pragma unroll
            for (int g = 0; g < 4; ++g)
                acc[f][g] = __builtin_amdgcn_wmma_f32_16x16x32_bf16(
                    false, a[f], false, b[g], (short)0, acc[f][g], false, false);
        __syncthreads();
    }

    const int nl = lane & 15, mh = lane >> 4;
    const float scale = 0.125f;                      // 1/sqrt(64)
#pragma unroll
    for (int f = 0; f < 2; ++f)
#pragma unroll
        for (int g = 0; g < 4; ++g) {
            const int n = n0 + waveN * 64 + g * 16 + nl;
#pragma unroll
            for (int r = 0; r < 8; ++r) {
                const int m = m0 + waveM * 32 + f * 16 + r + 8 * mh;
                out[(size_t)m * SEQ + n] = scale * acc[f][g][r];
            }
        }
}

// ---------------------------------------------------------------------------
// Row softmax in-place on the [B*H*S, 2048] score matrix. One wave per row.
// ---------------------------------------------------------------------------
__global__ __launch_bounds__(256)
void softmax_rows_kernel(float* __restrict__ attn)
{
    const int wave = threadIdx.x >> 5;
    const int lane = threadIdx.x & 31;
    const size_t row = (size_t)blockIdx.x * 8 + wave;
    float4* p4 = (float4*)(attn + row * SEQ);

    float4 v[16];
    float mx = -3.402823466e+38f;
#pragma unroll
    for (int i = 0; i < 16; ++i) {
        v[i] = p4[i * 32 + lane];
        mx = fmaxf(mx, fmaxf(fmaxf(v[i].x, v[i].y), fmaxf(v[i].z, v[i].w)));
    }
#pragma unroll
    for (int off = 16; off >= 1; off >>= 1)
        mx = fmaxf(mx, __shfl_xor(mx, off, 32));

    float sum = 0.f;
#pragma unroll
    for (int i = 0; i < 16; ++i) {
        v[i].x = __expf(v[i].x - mx);
        v[i].y = __expf(v[i].y - mx);
        v[i].z = __expf(v[i].z - mx);
        v[i].w = __expf(v[i].w - mx);
        sum += v[i].x + v[i].y + v[i].z + v[i].w;
    }
#pragma unroll
    for (int off = 16; off >= 1; off >>= 1)
        sum += __shfl_xor(sum, off, 32);

    const float inv = 1.0f / sum;
#pragma unroll
    for (int i = 0; i < 16; ++i) {
        v[i].x *= inv; v[i].y *= inv; v[i].z *= inv; v[i].w *= inv;
        p4[i * 32 + lane] = v[i];
    }
}

// ---------------------------------------------------------------------------
// attn_out[b,:,h*64..] = P[b,h] @ v[b,h] : M=2048, N=64, K=2048 per (b,h).
// P: fp32 from d_out, converted while staging; Vt: bf16 async-to-LDS.
// Double-buffered, wave tile 32x32 (2x2 frags).
// ---------------------------------------------------------------------------
__global__ __launch_bounds__(256)
void av_gemm_kernel(const float* __restrict__ attn, const bf16* __restrict__ vT,
                    bf16* __restrict__ attn_out)
{
    constexpr int BM = 128, BN = 64, BK = 32, LDK = BK + 16;
    __shared__ bf16 As[2][BM][LDK];
    __shared__ bf16 Bs[2][BN][LDK];

    const int bh = blockIdx.z;
    const int h  = bh & (N_HEADS - 1);
    const int b_ = bh >> 4;
    const float* P = attn + (size_t)bh * SEQ * SEQ;
    const bf16*  V = vT   + (size_t)bh * DK * SEQ;

    const int m0 = blockIdx.y * BM;
    const int t = threadIdx.x, lane = t & 31, wave = t >> 5;
    const int waveM = wave >> 1, waveN = wave & 1;
    const int lr = lane & 15, lk = (lane >> 4) << 4;

    v8f acc[2][2] = {};
    const int arow = t >> 1, akseg = (t & 1) << 4;     // A: 2 threads/row
    const int brow = t >> 2, bkseg = (t & 3) << 3;     // B: 4 threads/row, 8 elems

    auto stage = [&](int k0, int buf) {
        const float4* src = (const float4*)(P + (size_t)(m0 + arow) * SEQ + k0 + akseg);
#pragma unroll
        for (int q = 0; q < 4; ++q) {
            float4 x = src[q];
            As[buf][arow][akseg + q * 4 + 0] = (bf16)x.x;
            As[buf][arow][akseg + q * 4 + 1] = (bf16)x.y;
            As[buf][arow][akseg + q * 4 + 2] = (bf16)x.z;
            As[buf][arow][akseg + q * 4 + 3] = (bf16)x.w;
        }
        async_b128(&Bs[buf][brow][bkseg], V + (size_t)brow * SEQ + k0 + bkseg);
    };

    stage(0, 0);
    constexpr int NIT = SEQ / BK;                      // 64
    for (int it = 0; it < NIT; ++it) {
        const int cur = it & 1;
        const bool hasNext = (it + 1) < NIT;
        if (hasNext) {
            stage((it + 1) * BK, cur ^ 1);
            __builtin_prefetch(P + (size_t)(m0 + arow) * SEQ + (it + 1) * BK + akseg, 0, 1);
        }
        if (hasNext) wait_async_le<1>(); else wait_async_0();
        __syncthreads();

        v16bf a[2], b[2];
#pragma unroll
        for (int f = 0; f < 2; ++f)
            a[f] = *(const v16bf*)&As[cur][waveM * 32 + f * 16 + lr][lk];
#pragma unroll
        for (int g = 0; g < 2; ++g)
            b[g] = *(const v16bf*)&Bs[cur][waveN * 32 + g * 16 + lr][lk];
#pragma unroll
        for (int f = 0; f < 2; ++f)
#pragma unroll
            for (int g = 0; g < 2; ++g)
                acc[f][g] = __builtin_amdgcn_wmma_f32_16x16x32_bf16(
                    false, a[f], false, b[g], (short)0, acc[f][g], false, false);
        __syncthreads();
    }

    const int nl = lane & 15, mh = lane >> 4;
#pragma unroll
    for (int f = 0; f < 2; ++f)
#pragma unroll
        for (int g = 0; g < 2; ++g) {
            const int n = waveN * 32 + g * 16 + nl;    // 0..63 within head
#pragma unroll
            for (int r = 0; r < 8; ++r) {
                const int m = m0 + waveM * 32 + f * 16 + r + 8 * mh;
                attn_out[((size_t)b_ * SEQ + m) * D_MODEL + h * DK + n] =
                    (bf16)acc[f][g][r];
            }
        }
}

// ---------------------------------------------------------------------------
extern "C" void kernel_launch(void* const* d_in, const int* in_sizes, int n_in,
                              void* d_out, int out_size, void* d_ws, size_t ws_size,
                              hipStream_t stream) {
    const float* Q  = (const float*)d_in[0];
    const float* K  = (const float*)d_in[1];
    const float* V  = (const float*)d_in[2];
    const float* Wq = (const float*)d_in[3]; const float* bq = (const float*)d_in[4];
    const float* Wk = (const float*)d_in[5]; const float* bk = (const float*)d_in[6];
    const float* Wv = (const float*)d_in[7]; const float* bv = (const float*)d_in[8];
    const float* Wo = (const float*)d_in[9]; const float* bo = (const float*)d_in[10];

    float* out  = (float*)d_out;                              // [B,S,D] = 4096x1024
    float* attn = out + (size_t)M_TOK * D_MODEL;              // [B,H,S,S]

    const size_t headElems = (size_t)BATCH * N_HEADS * SEQ * DK;  // 4M elements
    bf16* qbf = (bf16*)d_ws;
    bf16* kbf = qbf + headElems;
    bf16* vT  = kbf + headElems;
    bf16* aob = vT  + headElems;                              // attn_out bf16 [B,S,D]

    dim3 blk(256);
    dim3 gproj(D_MODEL / 128, M_TOK / 128);                   // (8, 32)

    proj_gemm_kernel<0, 0><<<gproj, blk, 0, stream>>>(Q, Wq, bq, qbf);
    proj_gemm_kernel<0, 0><<<gproj, blk, 0, stream>>>(K, Wk, bk, kbf);
    proj_gemm_kernel<0, 1><<<gproj, blk, 0, stream>>>(V, Wv, bv, vT);

    dim3 gscore(SEQ / 128, SEQ / 128, BATCH * N_HEADS);       // (16,16,32)
    score_gemm_kernel<<<gscore, blk, 0, stream>>>(qbf, kbf, attn);

    softmax_rows_kernel<<<dim3((BATCH * N_HEADS * SEQ) / 8), blk, 0, stream>>>(attn);

    av_gemm_kernel<<<dim3(1, SEQ / 128, BATCH * N_HEADS), blk, 0, stream>>>(attn, vT, aob);

    proj_gemm_kernel<1, 2><<<gproj, blk, 0, stream>>>(aob, Wo, bo, out);
}
